// ModernTransformerBlock_64793876627795
// MI455X (gfx1250) — matrix operand order repaired and checked by reading, not compiled, not run on hardware
//
#include <hip/hip_runtime.h>
#include <hip/hip_bf16.h>

// ---------------------------------------------------------------------------
// ModernTransformerBlock for gfx1250 (CDNA5, wave32, WMMA).
// B=2, L=1024, E=1024, H=16, D=64.  All GEMMs via v_wmma_f32_16x16x32_bf16.
// ---------------------------------------------------------------------------

typedef __attribute__((ext_vector_type(16))) __bf16 v16bf;
typedef __attribute__((ext_vector_type(8)))  float  v8f;

union FragU { v16bf bf; uint4 q[2]; };

__device__ __forceinline__ unsigned short f2bf(float f) {
  union { float f; unsigned u; } v; v.f = f;
  unsigned r = v.u + 0x7FFFu + ((v.u >> 16) & 1u);   // round-to-nearest-even
  return (unsigned short)(r >> 16);
}

__device__ __forceinline__ v8f wmma_bf16(v16bf a, v16bf b, v8f c) {
  // D = A(16x32) * B(32x16) + C(16x16), f32 accumulate
  return __builtin_amdgcn_wmma_f32_16x16x32_bf16(false, a, false, b, (short)0, c,
                                                 false, false);
}

// A operand (16x32, bf16), row-major LDS tile with leading dim `ld` (multiple of 8).
// Lane m=lane&15, half=lane>>4.  elems 0-7 -> K = k0+8*half+0..7 ; elems 8-15 -> K+16.
__device__ __forceinline__ v16bf load_a_frag(const unsigned short* base, int ld,
                                             int row0, int k0, int lane) {
  const int m = lane & 15, half = lane >> 4;
  const unsigned short* p = base + (row0 + m) * ld + k0 + 8 * half;
  FragU f;
  f.q[0] = *(const uint4*)(p);
  f.q[1] = *(const uint4*)(p + 16);
  return f.bf;
}

// B operand (32x16, bf16): lane n=lane&15, half=lane>>4.  elems 0-7 -> K = k0+8*half+0..7 ; elems 8-15 -> K+16.
__device__ __forceinline__ v16bf load_b_frag(const unsigned short* base, int ld,
                                             int k0, int n0, int lane) {
  const int n = lane & 15, half = lane >> 4;
  const unsigned short* p = base + (k0 + 8 * half) * ld + n0 + n;
  union { v16bf bf; unsigned short s[16]; } f;
#pragma unroll
  for (int s = 0; s < 8; ++s) {
    f.s[s] = p[s * ld];
    f.s[s + 8] = p[(s + 16) * ld];
  }
  return f.bf;
}

// ---------------------------------------------------------------------------
// f32 -> bf16 bulk convert (weights), 4 elems/thread
// ---------------------------------------------------------------------------
__global__ __launch_bounds__(256) void f32_to_bf16_kernel(
    const float* __restrict__ in, unsigned short* __restrict__ out, int n4) {
  int i = blockIdx.x * blockDim.x + threadIdx.x;
  if (i < n4) {
    float4 v = ((const float4*)in)[i];
    uint2 p;
    p.x = (unsigned)f2bf(v.x) | ((unsigned)f2bf(v.y) << 16);
    p.y = (unsigned)f2bf(v.z) | ((unsigned)f2bf(v.w) << 16);
    ((uint2*)out)[i] = p;
  }
}

// ---------------------------------------------------------------------------
// RMSNorm: one block per row of 1024, output bf16
// ---------------------------------------------------------------------------
__global__ __launch_bounds__(256) void rmsnorm_bf16_kernel(
    const float* __restrict__ x, const float* __restrict__ w,
    unsigned short* __restrict__ out) {
  const int row = blockIdx.x;
  const int t = threadIdx.x;
  const float4 xv = ((const float4*)(x + (size_t)row * 1024))[t];
  float ss = xv.x * xv.x + xv.y * xv.y + xv.z * xv.z + xv.w * xv.w;
#pragma unroll
  for (int off = 16; off >= 1; off >>= 1) ss += __shfl_xor(ss, off, 32);
  __shared__ float warp_s[8];
  if ((t & 31) == 0) warp_s[t >> 5] = ss;
  __syncthreads();
  float tot = 0.0f;
#pragma unroll
  for (int i = 0; i < 8; ++i) tot += warp_s[i];
  const float scale = rsqrtf(tot * (1.0f / 1024.0f) + 1e-6f);
  const float4 wv = ((const float4*)w)[t];
  uint2 p;
  p.x = (unsigned)f2bf(xv.x * scale * wv.x) | ((unsigned)f2bf(xv.y * scale * wv.y) << 16);
  p.y = (unsigned)f2bf(xv.z * scale * wv.z) | ((unsigned)f2bf(xv.w * scale * wv.w) << 16);
  ((uint2*)(out + (size_t)row * 1024))[t] = p;
}

// ---------------------------------------------------------------------------
// bf16 WMMA GEMM:  C[M,N] = A[M,K] @ W[K,N] + bias (+residual / GELU)
// 128x128 tile, 64-wide K tile with register double-buffering.
// 256 threads = 8 waves (2x4); each wave owns 4x2 16x16 output tiles.
// MODE 0: outF = acc+bias               (QKV)
// MODE 1: outF = resid + acc+bias       (out-proj residual, MLP2 residual)
// MODE 2: outBf = bf16(gelu(acc+bias))  (MLP1)
// ---------------------------------------------------------------------------
template <int MODE>
__global__ __launch_bounds__(256) void gemm_bf16_kernel(
    const unsigned short* __restrict__ A, const unsigned short* __restrict__ Bw,
    const float* __restrict__ bias, const float* __restrict__ resid,
    float* __restrict__ outF, unsigned short* __restrict__ outBf,
    int M, int N, int K) {
  __shared__ unsigned short At[128][72];   // 64 K-cols + pad
  __shared__ unsigned short Bt[64][136];   // 128 N-cols + pad

  const int t = threadIdx.x;
  const int lane = t & 31;
  const int w = t >> 5;
  const int waveM = w >> 2, waveN = w & 3;
  const int m0 = blockIdx.y * 128;
  const int n0 = blockIdx.x * 128;

  v8f acc[4][2];
#pragma unroll
  for (int i = 0; i < 4; ++i)
#pragma unroll
    for (int j = 0; j < 2; ++j) acc[i][j] = (v8f){0, 0, 0, 0, 0, 0, 0, 0};

  // staging maps: A tile 128x64 ushort, B tile 64x128 ushort; 16B per thread per pass
  const int ar = t >> 3, ac = (t & 7) * 8;   // 32 rows/pass, 4 passes
  const int br = t >> 4, bc = (t & 15) * 8;  // 16 rows/pass, 4 passes

  uint4 aR[4], bR[4];
  // prologue: stage K-tile 0
#pragma unroll
  for (int p = 0; p < 4; ++p)
    aR[p] = *(const uint4*)(A + (size_t)(m0 + ar + 32 * p) * K + ac);
#pragma unroll
  for (int p = 0; p < 4; ++p)
    bR[p] = *(const uint4*)(Bw + (size_t)(br + 16 * p) * N + n0 + bc);

  for (int kk = 0; kk < K; kk += 64) {
    // commit staged registers to LDS
#pragma unroll
    for (int p = 0; p < 4; ++p) *(uint4*)&At[ar + 32 * p][ac] = aR[p];
#pragma unroll
    for (int p = 0; p < 4; ++p) *(uint4*)&Bt[br + 16 * p][bc] = bR[p];
    __syncthreads();

    // stage next K-tile while WMMAs run on the current one
    if (kk + 64 < K) {
#pragma unroll
      for (int p = 0; p < 4; ++p)
        aR[p] = *(const uint4*)(A + (size_t)(m0 + ar + 32 * p) * K + kk + 64 + ac);
#pragma unroll
      for (int p = 0; p < 4; ++p)
        bR[p] = *(const uint4*)(Bw + (size_t)(kk + 64 + br + 16 * p) * N + n0 + bc);
    }

#pragma unroll
    for (int ks = 0; ks < 2; ++ks) {
      v16bf bfr[2];
#pragma unroll
      for (int nt = 0; nt < 2; ++nt)
        bfr[nt] = load_b_frag(&Bt[0][0], 136, ks * 32, waveN * 32 + nt * 16, lane);
#pragma unroll
      for (int mi = 0; mi < 4; ++mi) {
        v16bf afr = load_a_frag(&At[0][0], 72, waveM * 64 + mi * 16, ks * 32, lane);
#pragma unroll
        for (int nt = 0; nt < 2; ++nt)
          acc[mi][nt] = wmma_bf16(afr, bfr[nt], acc[mi][nt]);
      }
    }
    __syncthreads();
  }

  // epilogue: C layout -> lane holds (M = r + 8*(lane>>4), N = lane&15)
  const int half = lane >> 4, nlo = lane & 15;
#pragma unroll
  for (int mi = 0; mi < 4; ++mi) {
#pragma unroll
    for (int nt = 0; nt < 2; ++nt) {
      const int col = n0 + waveN * 32 + nt * 16 + nlo;
      const float bcol = bias[col];
#pragma unroll
      for (int r = 0; r < 8; ++r) {
        const int row = m0 + waveM * 64 + mi * 16 + r + 8 * half;
        const size_t off = (size_t)row * N + col;
        float v = acc[mi][nt][r] + bcol;
        if (MODE == 0) {
          outF[off] = v;
        } else if (MODE == 1) {
          outF[off] = resid[off] + v;
        } else {  // gelu (tanh approx) -> bf16
          float g = 0.5f * v *
                    (1.0f + tanhf(0.7978845608f * (v + 0.044715f * v * v * v)));
          outBf[off] = f2bf(g);
        }
      }
    }
  }
}

// ---------------------------------------------------------------------------
// Cosformer causal linear attention, chunked (chunk=64), all-WMMA.
// One workgroup (128 threads = 4 waves) per (b,h).
// Feature concat: Qp/Kp = [relu(q)*cos | relu(q)*sin]  (64 x 128).
// V augmented with a ones column (N=80) so ctx[:,64] = normalizer.
//   per chunk:  S = Qp @ KpT          (causal-masked)
//               ctx = S@Vaug + Qp@KV_state
//               KV_state += KpT@Vaug  (persistent f32 regs; bf16 copy in LDS)
// ---------------------------------------------------------------------------
__global__ __launch_bounds__(128) void cosformer_attn_kernel(
    const float* __restrict__ qkv, unsigned short* __restrict__ outBf) {
  extern __shared__ unsigned short sm[];
  unsigned short* Qp  = sm;                 // [64][136]  (64x128 + pad)
  unsigned short* KpT = Qp + 64 * 136;      // [128][72]  (128x64 + pad)
  unsigned short* Vb  = KpT + 128 * 72;     // [64][88]   (64x80 + pad)
  unsigned short* Sb  = Vb + 64 * 88;       // [64][72]   (64x64 + pad)
  unsigned short* KVb = Sb + 64 * 72;       // [128][88]  (128x80 + pad)

  const int t = threadIdx.x, lane = t & 31, w = t >> 5;
  const int bh = blockIdx.x, b = bh >> 4, h = bh & 15;
  const int half = lane >> 4, nlo = lane & 15;

  // init: KV state bf16 copy = 0; V augmentation columns (col 64 = 1, 65..79 = 0)
  for (int i = t; i < 128 * 88; i += 128) KVb[i] = 0;
  for (int i = t; i < 64 * 16; i += 128) {
    int jr = i >> 4, c = i & 15;
    Vb[jr * 88 + 64 + c] = (c == 0) ? (unsigned short)0x3F80 : (unsigned short)0;
  }

  v8f kvAcc[2][5];
#pragma unroll
  for (int s2 = 0; s2 < 2; ++s2)
#pragma unroll
    for (int nt = 0; nt < 5; ++nt) kvAcc[s2][nt] = (v8f){0, 0, 0, 0, 0, 0, 0, 0};

  const int jrow = t >> 1;            // chunk-local row this thread produces
  const int dbase = (t & 1) * 32;     // half of the 64 feature dims

  for (int ct = 0; ct < 16; ++ct) {
    __syncthreads();  // previous consumers done; KVb/init visible

    // ---- produce Qp, KpT, Vb for this chunk ----
    {
      const int l = ct * 64 + jrow;
      const size_t row = (size_t)b * 1024 + l;
      const float ang = 1.5707963268f * (float)l * (1.0f / 1024.0f);
      const float cw = cosf(ang), sw = sinf(ang);
      const float* qp = qkv + row * 3072 + h * 64;
#pragma unroll
      for (int dd = 0; dd < 32; ++dd) {
        const int d = dbase + dd;
        float q = fmaxf(qp[d], 0.0f);
        float k = fmaxf(qp[1024 + d], 0.0f);
        float v = qp[2048 + d];
        Qp[jrow * 136 + d]        = f2bf(q * cw);
        Qp[jrow * 136 + 64 + d]   = f2bf(q * sw);
        KpT[d * 72 + jrow]        = f2bf(k * cw);
        KpT[(64 + d) * 72 + jrow] = f2bf(k * sw);
        Vb[jrow * 88 + d]         = f2bf(v);
      }
    }
    __syncthreads();

    // Q fragments for this wave's 16-row strip, K=128
    v16bf qa[4];
#pragma unroll
    for (int kk = 0; kk < 4; ++kk)
      qa[kk] = load_a_frag(Qp, 136, 16 * w, 32 * kk, lane);

    // ---- S = Qp @ KpT, causal mask, store bf16 ----
#pragma unroll
    for (int nt = 0; nt < 4; ++nt) {
      v8f s = (v8f){0, 0, 0, 0, 0, 0, 0, 0};
#pragma unroll
      for (int kk = 0; kk < 4; ++kk)
        s = wmma_bf16(qa[kk], load_b_frag(KpT, 72, 32 * kk, 16 * nt, lane), s);
#pragma unroll
      for (int r = 0; r < 8; ++r) {
        const int i = 16 * w + r + 8 * half;
        const int jj = 16 * nt + nlo;
        Sb[i * 72 + jj] = (jj <= i) ? f2bf(s[r]) : (unsigned short)0;
      }
    }
    __syncthreads();  // Sb ready; prev-chunk KVb still intact

    // ---- ctx = Qp@KVb (inter) + Sb@Vaug (intra) ----
    v16bf sa[2];
    sa[0] = load_a_frag(Sb, 72, 16 * w, 0, lane);
    sa[1] = load_a_frag(Sb, 72, 16 * w, 32, lane);
    v8f ctx[5];
#pragma unroll
    for (int nt = 0; nt < 5; ++nt) {
      v8f a = (v8f){0, 0, 0, 0, 0, 0, 0, 0};
#pragma unroll
      for (int kk = 0; kk < 4; ++kk)
        a = wmma_bf16(qa[kk], load_b_frag(KVb, 88, 32 * kk, 16 * nt, lane), a);
#pragma unroll
      for (int kk = 0; kk < 2; ++kk)
        a = wmma_bf16(sa[kk], load_b_frag(Vb, 88, 32 * kk, 16 * nt, lane), a);
      ctx[nt] = a;
    }

    // ---- divide by normalizer (ctx tile 4, col 0 lives in lanes 0/16) ----
#pragma unroll
    for (int r = 0; r < 8; ++r) {
      const float nv = __shfl(ctx[4][r], half * 16, 32);
      const float inv = 1.0f / (nv + 1e-6f);
      const int i = 16 * w + r + 8 * half;
      const size_t row = (size_t)b * 1024 + ct * 64 + i;
#pragma unroll
      for (int nt = 0; nt < 4; ++nt)
        outBf[row * 1024 + h * 64 + 16 * nt + nlo] = f2bf(ctx[nt][r] * inv);
    }

    // ---- state update: KV += KpT @ Vaug (persistent f32 accumulators) ----
    v16bf ka0[2], ka1[2];
    ka0[0] = load_a_frag(KpT, 72, 16 * w, 0, lane);
    ka0[1] = load_a_frag(KpT, 72, 16 * w, 32, lane);
    ka1[0] = load_a_frag(KpT, 72, 16 * (w + 4), 0, lane);
    ka1[1] = load_a_frag(KpT, 72, 16 * (w + 4), 32, lane);
#pragma unroll
    for (int nt = 0; nt < 5; ++nt) {
      v16bf b0 = load_b_frag(Vb, 88, 0, 16 * nt, lane);
      v16bf b1 = load_b_frag(Vb, 88, 32, 16 * nt, lane);
      kvAcc[0][nt] = wmma_bf16(ka0[0], b0, kvAcc[0][nt]);
      kvAcc[0][nt] = wmma_bf16(ka0[1], b1, kvAcc[0][nt]);
      kvAcc[1][nt] = wmma_bf16(ka1[0], b0, kvAcc[1][nt]);
      kvAcc[1][nt] = wmma_bf16(ka1[1], b1, kvAcc[1][nt]);
    }
    __syncthreads();  // everyone done reading KVb

    // write bf16 copy of updated state for next chunk's B operand
#pragma unroll
    for (int s2 = 0; s2 < 2; ++s2)
#pragma unroll
      for (int nt = 0; nt < 5; ++nt)
#pragma unroll
        for (int r = 0; r < 8; ++r) {
          const int rw = 16 * (w + 4 * s2) + r + 8 * half;
          KVb[rw * 88 + 16 * nt + nlo] = f2bf(kvAcc[s2][nt][r]);
        }
  }
}

// ---------------------------------------------------------------------------
// Host-side launch sequence
// ---------------------------------------------------------------------------
extern "C" void kernel_launch(void* const* d_in, const int* in_sizes, int n_in,
                              void* d_out, int out_size, void* d_ws, size_t ws_size,
                              hipStream_t stream) {
  (void)in_sizes; (void)n_in; (void)out_size; (void)ws_size;
  const float* x       = (const float*)d_in[0];
  const float* qkv_w   = (const float*)d_in[1];
  const float* qkv_b   = (const float*)d_in[2];
  const float* out_w   = (const float*)d_in[3];
  const float* out_b   = (const float*)d_in[4];
  const float* norm1_w = (const float*)d_in[5];
  const float* norm2_w = (const float*)d_in[6];
  const float* mlp_w1  = (const float*)d_in[7];
  const float* mlp_b1  = (const float*)d_in[8];
  const float* mlp_w2  = (const float*)d_in[9];
  const float* mlp_b2  = (const float*)d_in[10];
  float* out = (float*)d_out;

  const int M = 2048, E = 1024;

  char* p = (char*)d_ws;
  auto carve = [&](size_t bytes) {
    char* r = p;
    p += (bytes + 255) & ~(size_t)255;
    return r;
  };
  unsigned short* qkvw_bf = (unsigned short*)carve((size_t)E * 3072 * 2);
  unsigned short* outw_bf = (unsigned short*)carve((size_t)E * 1024 * 2);
  unsigned short* w1_bf   = (unsigned short*)carve((size_t)E * 4096 * 2);
  unsigned short* w2_bf   = (unsigned short*)carve((size_t)4096 * 1024 * 2);
  unsigned short* xn_bf   = (unsigned short*)carve((size_t)M * E * 2);
  float*          qkv_f   = (float*)carve((size_t)M * 3072 * 4);
  unsigned short* attn_bf = (unsigned short*)carve((size_t)M * E * 2);
  float*          h_f     = (float*)carve((size_t)M * E * 4);
  unsigned short* h2n_bf  = (unsigned short*)carve((size_t)M * E * 2);
  unsigned short* mid_bf  = (unsigned short*)carve((size_t)M * 4096 * 2);

  // weight conversions f32 -> bf16
  f32_to_bf16_kernel<<<(E * 3072 / 4 + 255) / 256, 256, 0, stream>>>(qkv_w, qkvw_bf, E * 3072 / 4);
  f32_to_bf16_kernel<<<(E * 1024 / 4 + 255) / 256, 256, 0, stream>>>(out_w, outw_bf, E * 1024 / 4);
  f32_to_bf16_kernel<<<(E * 4096 / 4 + 255) / 256, 256, 0, stream>>>(mlp_w1, w1_bf, E * 4096 / 4);
  f32_to_bf16_kernel<<<(4096 * 1024 / 4 + 255) / 256, 256, 0, stream>>>(mlp_w2, w2_bf, 4096 * 1024 / 4);

  // norm1 -> bf16
  rmsnorm_bf16_kernel<<<M, 256, 0, stream>>>(x, norm1_w, xn_bf);

  // QKV projection: [2048,1024] @ [1024,3072] + bias -> f32
  gemm_bf16_kernel<0><<<dim3(3072 / 128, M / 128), 256, 0, stream>>>(
      xn_bf, qkvw_bf, qkv_b, nullptr, qkv_f, nullptr, M, 3072, 1024);

  // cosformer attention -> bf16 [2048,1024]
  const int attn_smem = (64 * 136 + 128 * 72 + 64 * 88 + 64 * 72 + 128 * 88) * 2;  // 78848 B
  (void)hipFuncSetAttribute((const void*)cosformer_attn_kernel,
                            hipFuncAttributeMaxDynamicSharedMemorySize, attn_smem);
  cosformer_attn_kernel<<<32, 128, attn_smem, stream>>>(qkv_f, attn_bf);

  // out projection + residual: h = x + attn@out_w + out_b
  gemm_bf16_kernel<1><<<dim3(1024 / 128, M / 128), 256, 0, stream>>>(
      attn_bf, outw_bf, out_b, x, h_f, nullptr, M, 1024, 1024);

  // norm2 -> bf16
  rmsnorm_bf16_kernel<<<M, 256, 0, stream>>>(h_f, norm2_w, h2n_bf);

  // MLP up + GELU -> bf16
  gemm_bf16_kernel<2><<<dim3(4096 / 128, M / 128), 256, 0, stream>>>(
      h2n_bf, w1_bf, mlp_b1, nullptr, nullptr, mid_bf, M, 4096, 1024);

  // MLP down + residual: out = h + mid@w2 + b2
  gemm_bf16_kernel<1><<<dim3(1024 / 128, M / 128), 256, 0, stream>>>(
      mid_bf, w2_bf, mlp_b2, h_f, out, nullptr, M, 1024, 4096);
}
